// EncoderRNN_32263794328077
// MI455X (gfx1250) — matrix operand order repaired and checked by reading, not compiled, not run on hardware
//
#include <hip/hip_runtime.h>
#include <hip/hip_bf16.h>

// ---------------------------------------------------------------------------
// EncoderRNN for MI455X (gfx1250): bf16 WMMA GEMMs + in-register LSTM math.
// K-step = 64 with double-buffered LDS B-tiles: 8 v_wmma per single barrier.
// ---------------------------------------------------------------------------

typedef __bf16 bf16_t;
typedef __bf16 v16bf __attribute__((ext_vector_type(16)));
typedef float  v8f   __attribute__((ext_vector_type(8)));

#define BB 128   // batch
#define TT 64    // time steps
#define FF 2048  // input features
#define HH 1024  // hidden
#define MMD 512  // boundary dim

#define BPAD 72  // LDS row stride in bf16 (144B: 16B aligned, staggered banks)

union FragAB { v16bf v; uint4 q[2]; };

__device__ __forceinline__ float sigmoidf_(float x) { return 1.0f / (1.0f + __expf(-x)); }

// ---------------------------------------------------------------------------
// Elementwise helpers
// ---------------------------------------------------------------------------
__global__ void k_f32_to_bf16(const float* __restrict__ src, bf16_t* __restrict__ dst, int n) {
    for (int i = blockIdx.x * blockDim.x + threadIdx.x; i < n; i += gridDim.x * blockDim.x)
        dst[i] = (bf16_t)src[i];
}

// dst[n][k] = k < kh ? Wih[n][k] : Whh[n][k-kh]   (bf16), dst is [nrows][2*kh]
__global__ void k_concat_w(const float* __restrict__ wih, const float* __restrict__ whh,
                           bf16_t* __restrict__ dst, int nrows, int kh) {
    int total = nrows * 2 * kh;
    for (int i = blockIdx.x * blockDim.x + threadIdx.x; i < total; i += gridDim.x * blockDim.x) {
        int n = i / (2 * kh);
        int k = i - n * (2 * kh);
        float v = (k < kh) ? wih[n * kh + k] : whh[n * kh + (k - kh)];
        dst[i] = (bf16_t)v;
    }
}

__global__ void k_zero_f32(float* __restrict__ p, int n) {
    for (int i = blockIdx.x * blockDim.x + threadIdx.x; i < n; i += gridDim.x * blockDim.x)
        p[i] = 0.0f;
}
__global__ void k_zero_bf16(bf16_t* __restrict__ p, int n) {
    for (int i = blockIdx.x * blockDim.x + threadIdx.x; i < n; i += gridDim.x * blockDim.x)
        p[i] = (bf16_t)0.0f;
}

// Boundary detector precompute: wsx = Wsi^T vs, wsh = Wsh^T vs, b0 = b_bd . vs
__global__ void k_bd_precompute(const float* __restrict__ Wsi, const float* __restrict__ Wsh,
                                const float* __restrict__ vs, const float* __restrict__ bbd,
                                float* __restrict__ wsx, float* __restrict__ wshv,
                                float* __restrict__ b0) {
    int h = blockIdx.x * blockDim.x + threadIdx.x;
    if (h < HH) {
        float a = 0.0f, b = 0.0f;
        for (int m = 0; m < MMD; ++m) {
            float v = vs[m];
            a += Wsi[m * HH + h] * v;
            b += Wsh[m * HH + h] * v;
        }
        wsx[h] = a;
        wshv[h] = b;
    }
    if (blockIdx.x == 0 && threadIdx.x == 0) {
        float s = 0.0f;
        for (int m = 0; m < MMD; ++m) s += bbd[m] * vs[m];
        *b0 = s;
    }
}

// Per-step boundary gate: s[b] = round(sigmoid(x.wsx + h.wsh + b0)) = (pre >= 0)
__global__ void k_boundary(const bf16_t* __restrict__ x, const bf16_t* __restrict__ h1,
                           const float* __restrict__ wsx, const float* __restrict__ wshv,
                           const float* __restrict__ b0, float* __restrict__ s) {
    int b = threadIdx.x;  // 128 threads
    if (b < BB) {
        float acc = *b0;
        const bf16_t* xr = x + (size_t)b * HH;
        const bf16_t* hr = h1 + (size_t)b * HH;
        for (int k = 0; k < HH; ++k)
            acc += (float)xr[k] * wsx[k] + (float)hr[k] * wshv[k];
        s[b] = (acc >= 0.0f) ? 1.0f : 0.0f;
    }
}

// ---------------------------------------------------------------------------
// Shared tile helpers: stage one 64(N) x 64(K) bf16 tile into LDS buffer.
// 256 threads: 4 threads per N-row, 16 bf16 (2 x b128) per thread.
// ---------------------------------------------------------------------------
__device__ __forceinline__ void stage_tile(bf16_t (*Bs)[BPAD], const bf16_t* wrow_k0,
                                           int ldrow, int ldch) {
    const uint4* g = reinterpret_cast<const uint4*>(wrow_k0 + ldch);
    uint4 q0 = g[0];
    uint4 q1 = g[1];
    *reinterpret_cast<uint4*>(&Bs[ldrow][ldch]) = q0;
    *reinterpret_cast<uint4*>(&Bs[ldrow][ldch + 8]) = q1;
}

// ---------------------------------------------------------------------------
// Embedding GEMM: out[t][b][h] = relu(X[b*T+t] . Wemb[h] + bias[h]), bf16 out.
// Block = 8 waves; wave w owns M rows [by*128 + 16w, +16); block owns 64 N cols.
// ---------------------------------------------------------------------------
__global__ __launch_bounds__(256) void k_embed_gemm(const bf16_t* __restrict__ X,
                                                    const bf16_t* __restrict__ W,
                                                    const float* __restrict__ bias,
                                                    bf16_t* __restrict__ out) {
    __shared__ bf16_t Bs[2][64][BPAD];
    const int tid = threadIdx.x;
    const int wave = tid >> 5, lane = tid & 31;
    const int n0 = blockIdx.x * 64;
    const int m0 = blockIdx.y * 128 + wave * 16;
    const int mrow = m0 + (lane & 15);
    const bf16_t* arow = X + (size_t)mrow * FF;
    const int kb = (lane >> 4) * 8;    // A-frag K sub-chunk base
    const int kf = (lane >> 4) * 8;    // B-frag K base within 32-window
    const int ldrow = tid >> 2, ldch = (tid & 3) * 16;
    const bf16_t* wrow = W + (size_t)(n0 + ldrow) * FF;

    v8f acc[4] = {};

    stage_tile(Bs[0], wrow, ldrow, ldch);
    __syncthreads();

    const int NK = FF / 64;  // 32 iterations
    for (int it = 0; it < NK; ++it) {
        const int k0 = it * 64;
        const int cur = it & 1;

        FragAB a0, a1;
        a0.q[0] = *reinterpret_cast<const uint4*>(arow + k0 + kb);
        a0.q[1] = *reinterpret_cast<const uint4*>(arow + k0 + kb + 16);
        a1.q[0] = *reinterpret_cast<const uint4*>(arow + k0 + 32 + kb);
        a1.q[1] = *reinterpret_cast<const uint4*>(arow + k0 + 32 + kb + 16);

        if (it + 1 < NK) {
            __builtin_prefetch(wrow + k0 + 128, 0, 1);
            stage_tile(Bs[cur ^ 1], wrow + k0 + 64, ldrow, ldch);
        }

#pragma unroll
        for (int j = 0; j < 2; ++j) {
            const FragAB& a = j ? a1 : a0;
#pragma unroll
            for (int t4 = 0; t4 < 4; ++t4) {
                FragAB b;
                const uint4* p = reinterpret_cast<const uint4*>(
                    &Bs[cur][t4 * 16 + (lane & 15)][j * 32 + kf]);
                b.q[0] = p[0];
                b.q[1] = p[2];
                acc[t4] = __builtin_amdgcn_wmma_f32_16x16x32_bf16(false, a.v, false, b.v,
                                                                  (short)0, acc[t4], false,
                                                                  false);
            }
        }
        __syncthreads();
    }

    const int mhi = (lane >> 4) * 8;
#pragma unroll
    for (int t4 = 0; t4 < 4; ++t4) {
        int n = n0 + t4 * 16 + (lane & 15);
        float bv = bias[n];
#pragma unroll
        for (int r = 0; r < 8; ++r) {
            int m = m0 + r + mhi;           // flat [B,T] row
            int bs = m >> 6, tt = m & 63;   // b = m/T, t = m%T
            float v = acc[t4][r] + bv;
            v = v > 0.0f ? v : 0.0f;
            out[((size_t)tt * BB + bs) * HH + n] = (bf16_t)v;
        }
    }
}

// ---------------------------------------------------------------------------
// LSTM step GEMM + in-register gate math.
// gates[m][g*H+hid] = [x|h][m] . Wcat[g*H+hid]  (K = 2H = 2048)
// Block owns hid slice [hid0, hid0+16) across all 4 gates (64 N cols);
// wave w owns M rows [16w, 16w+16) -> acc[0..3] = i,f,g,o for same (m,hid).
// mode 1 (lstm1): +bias, boundary scaling via s; writes u = h*s, h_out = h*(1-s),
//                 c *= (1-s) in place.
// mode 2 (lstm2): no bias; writes h_out (bf16) and out_f32 (final h2).
// ---------------------------------------------------------------------------
__global__ __launch_bounds__(256) void k_lstm_step(const bf16_t* __restrict__ xA,
                                                   const bf16_t* __restrict__ hA,
                                                   const bf16_t* __restrict__ W,
                                                   const float* __restrict__ bias,
                                                   const float* __restrict__ s,
                                                   float* __restrict__ c,
                                                   bf16_t* __restrict__ out_u,
                                                   bf16_t* __restrict__ out_h,
                                                   float* __restrict__ out_f32, int mode) {
    __shared__ bf16_t Bs[2][64][BPAD];
    const int tid = threadIdx.x;
    const int wave = tid >> 5, lane = tid & 31;
    const int hid0 = blockIdx.x * 16;
    const int m0 = wave * 16;
    const int mrow = m0 + (lane & 15);
    const int kb = (lane >> 4) * 8;
    const int kf = (lane >> 4) * 8;
    const int ldrow = tid >> 2, ldch = (tid & 3) * 16;
    const int ldgate = ldrow >> 4, ldj = ldrow & 15;
    const size_t ldw = 2 * HH;
    const bf16_t* wrow = W + (size_t)(ldgate * HH + hid0 + ldj) * ldw;

    v8f acc[4] = {};

    stage_tile(Bs[0], wrow, ldrow, ldch);
    __syncthreads();

    const int NK = (2 * HH) / 64;  // 32 iterations
    for (int it = 0; it < NK; ++it) {
        const int k0 = it * 64;
        const int cur = it & 1;

        // 64-K window lies entirely in one half (64 | 1024)
        const bf16_t* arow = (k0 < HH) ? (xA + (size_t)mrow * HH + k0)
                                       : (hA + (size_t)mrow * HH + (k0 - HH));
        FragAB a0, a1;
        a0.q[0] = *reinterpret_cast<const uint4*>(arow + kb);
        a0.q[1] = *reinterpret_cast<const uint4*>(arow + kb + 16);
        a1.q[0] = *reinterpret_cast<const uint4*>(arow + 32 + kb);
        a1.q[1] = *reinterpret_cast<const uint4*>(arow + 32 + kb + 16);

        if (it + 1 < NK) {
            __builtin_prefetch(wrow + k0 + 128, 0, 1);
            stage_tile(Bs[cur ^ 1], wrow + k0 + 64, ldrow, ldch);
        }

#pragma unroll
        for (int j = 0; j < 2; ++j) {
            const FragAB& a = j ? a1 : a0;
#pragma unroll
            for (int g4 = 0; g4 < 4; ++g4) {
                FragAB b;
                const uint4* p = reinterpret_cast<const uint4*>(
                    &Bs[cur][g4 * 16 + (lane & 15)][j * 32 + kf]);
                b.q[0] = p[0];
                b.q[1] = p[2];
                acc[g4] = __builtin_amdgcn_wmma_f32_16x16x32_bf16(false, a.v, false, b.v,
                                                                  (short)0, acc[g4], false,
                                                                  false);
            }
        }
        __syncthreads();
    }

    const int mhi = (lane >> 4) * 8;
    const int hid = hid0 + (lane & 15);
    float bi = 0.0f, bf = 0.0f, bg = 0.0f, bo = 0.0f;
    if (mode == 1) {
        bi = bias[hid];
        bf = bias[HH + hid];
        bg = bias[2 * HH + hid];
        bo = bias[3 * HH + hid];
    }
#pragma unroll
    for (int r = 0; r < 8; ++r) {
        int m = m0 + r + mhi;
        size_t idx = (size_t)m * HH + hid;
        float iv = acc[0][r] + bi;
        float fv = acc[1][r] + bf;
        float gv = acc[2][r] + bg;
        float ov = acc[3][r] + bo;
        float cold = c[idx];
        float cn = sigmoidf_(fv) * cold + sigmoidf_(iv) * tanhf(gv);
        float hn = sigmoidf_(ov) * tanhf(cn);
        if (mode == 1) {
            float sv = s[m];
            out_u[idx] = (bf16_t)(hn * sv);
            out_h[idx] = (bf16_t)(hn * (1.0f - sv));
            c[idx] = cn * (1.0f - sv);
        } else {
            out_h[idx] = (bf16_t)hn;
            c[idx] = cn;
            out_f32[idx] = hn;
        }
    }
}

// ---------------------------------------------------------------------------
// Host-side launch
// ---------------------------------------------------------------------------
static inline void* carve(uintptr_t& p, size_t bytes) {
    void* r = (void*)p;
    p += (bytes + 255) & ~(size_t)255;
    return r;
}

extern "C" void kernel_launch(void* const* d_in, const int* in_sizes, int n_in,
                              void* d_out, int out_size, void* d_ws, size_t ws_size,
                              hipStream_t stream) {
    (void)in_sizes; (void)n_in; (void)out_size; (void)ws_size;

    const float* video  = (const float*)d_in[0];   // [B,T,F]
    const float* Wembed = (const float*)d_in[1];   // [H,F]
    const float* bembed = (const float*)d_in[2];   // [H]
    const float* Wih1   = (const float*)d_in[3];   // [4H,H]
    const float* Whh1   = (const float*)d_in[4];   // [4H,H]
    const float* b1     = (const float*)d_in[5];   // [4H]
    const float* Wsi    = (const float*)d_in[6];   // [M,H]
    const float* Wsh    = (const float*)d_in[7];   // [M,H]
    const float* bbd    = (const float*)d_in[8];   // [M]
    const float* vs     = (const float*)d_in[9];   // [M]
    const float* Wih2   = (const float*)d_in[10];  // [4H,H]
    const float* Whh2   = (const float*)d_in[11];  // [4H,H]
    float* out = (float*)d_out;                    // [B,H]

    uintptr_t p = (uintptr_t)d_ws;
    bf16_t* Xin_bf  = (bf16_t*)carve(p, (size_t)BB * TT * FF * sizeof(bf16_t));
    bf16_t* Wemb_bf = (bf16_t*)carve(p, (size_t)HH * FF * sizeof(bf16_t));
    bf16_t* Wcat1   = (bf16_t*)carve(p, (size_t)4 * HH * 2 * HH * sizeof(bf16_t));
    bf16_t* Wcat2   = (bf16_t*)carve(p, (size_t)4 * HH * 2 * HH * sizeof(bf16_t));
    bf16_t* Xbf     = (bf16_t*)carve(p, (size_t)TT * BB * HH * sizeof(bf16_t));
    bf16_t* h1A     = (bf16_t*)carve(p, (size_t)BB * HH * sizeof(bf16_t));
    bf16_t* h1Bb    = (bf16_t*)carve(p, (size_t)BB * HH * sizeof(bf16_t));
    bf16_t* h2A     = (bf16_t*)carve(p, (size_t)BB * HH * sizeof(bf16_t));
    bf16_t* h2Bb    = (bf16_t*)carve(p, (size_t)BB * HH * sizeof(bf16_t));
    bf16_t* u_bf    = (bf16_t*)carve(p, (size_t)BB * HH * sizeof(bf16_t));
    float*  c1      = (float*)carve(p, (size_t)BB * HH * sizeof(float));
    float*  c2      = (float*)carve(p, (size_t)BB * HH * sizeof(float));
    float*  h2dummy = (float*)carve(p, (size_t)BB * HH * sizeof(float));
    float*  wsx     = (float*)carve(p, (size_t)HH * sizeof(float));
    float*  wshv    = (float*)carve(p, (size_t)HH * sizeof(float));
    float*  b0      = (float*)carve(p, 256);
    float*  sbuf    = (float*)carve(p, (size_t)BB * sizeof(float));

    // --- one-time (per launch) conversions / precompute ---
    k_f32_to_bf16<<<4096, 256, 0, stream>>>(video, Xin_bf, BB * TT * FF);
    k_f32_to_bf16<<<2048, 256, 0, stream>>>(Wembed, Wemb_bf, HH * FF);
    k_concat_w<<<4096, 256, 0, stream>>>(Wih1, Whh1, Wcat1, 4 * HH, HH);
    k_concat_w<<<4096, 256, 0, stream>>>(Wih2, Whh2, Wcat2, 4 * HH, HH);
    k_bd_precompute<<<(HH + 255) / 256, 256, 0, stream>>>(Wsi, Wsh, vs, bbd, wsx, wshv, b0);
    k_zero_bf16<<<128, 256, 0, stream>>>(h1A, BB * HH);
    k_zero_bf16<<<128, 256, 0, stream>>>(h2A, BB * HH);
    k_zero_f32<<<128, 256, 0, stream>>>(c1, BB * HH);
    k_zero_f32<<<128, 256, 0, stream>>>(c2, BB * HH);

    // --- embedding GEMM (M=8192, N=1024, K=2048) ---
    k_embed_gemm<<<dim3(HH / 64, (BB * TT) / 128), 256, 0, stream>>>(Xin_bf, Wemb_bf, bembed,
                                                                     Xbf);

    // --- recurrent scan over T ---
    const bf16_t* h1_in = h1A;  bf16_t* h1_out = h1Bb;
    const bf16_t* h2_in = h2A;  bf16_t* h2_out = h2Bb;
    for (int t = 0; t < TT; ++t) {
        const bf16_t* x_t = Xbf + (size_t)t * BB * HH;
        k_boundary<<<1, 128, 0, stream>>>(x_t, h1_in, wsx, wshv, b0, sbuf);
        k_lstm_step<<<HH / 16, 256, 0, stream>>>(x_t, h1_in, Wcat1, b1, sbuf, c1,
                                                 u_bf, h1_out, nullptr, 1);
        float* f32dst = (t == TT - 1) ? out : h2dummy;
        k_lstm_step<<<HH / 16, 256, 0, stream>>>(u_bf, h2_in, Wcat2, nullptr, nullptr, c2,
                                                 nullptr, h2_out, f32dst, 2);
        // ping-pong
        const bf16_t* tmp1 = h1_in; h1_in = h1_out; h1_out = (bf16_t*)tmp1;
        const bf16_t* tmp2 = h2_in; h2_in = h2_out; h2_out = (bf16_t*)tmp2;
    }
}
